// Head_10101763080424
// MI455X (gfx1250) — matrix operand order repaired and checked by reading, not compiled, not run on hardware
//
#include <hip/hip_runtime.h>
#include <hip/hip_bf16.h>

typedef __attribute__((ext_vector_type(16))) _Float16 v16h;
typedef __attribute__((ext_vector_type(8)))  _Float16 v8h;
typedef __attribute__((ext_vector_type(8)))  float    v8f;

#define B_  8
#define T_  2048
#define C_  1024
#define H_  64

__device__ __forceinline__ v16h cvt2(v8f a, v8f b) {
  v16h r;
#pragma unroll
  for (int i = 0; i < 8; ++i) { r[i] = (_Float16)a[i]; r[8 + i] = (_Float16)b[i]; }
  return r;
}

__device__ __forceinline__ v16h combine8(v8h lo, v8h hi) {
  v16h r;
#pragma unroll
  for (int i = 0; i < 8; ++i) { r[i] = lo[i]; r[8 + i] = hi[i]; }
  return r;
}

// ---------------------------------------------------------------------------
// Phase 0: one-time W f32 -> f16 conversion (3 x 64 x 1024 = 196608 elements).
// W f16 (384 KB) stays L2-resident and is re-read cheaply by all phase-1 waves.
// ---------------------------------------------------------------------------
__global__ __launch_bounds__(256) void cvt_w_kernel(
    const float* __restrict__ Wk, const float* __restrict__ Wq,
    const float* __restrict__ Wv, _Float16* __restrict__ w16)
{
  const int i = blockIdx.x * 256 + threadIdx.x;     // [0, 65536)
  const int n = H_ * C_;
  w16[i]         = (_Float16)Wq[i];
  w16[n + i]     = (_Float16)Wk[i];
  w16[2 * n + i] = (_Float16)Wv[i];
}

// ---------------------------------------------------------------------------
// Phase 1: fused QKV projection.  x[16384,1024] f32  x  W16[64,1024] f16 (x3)
// -> q16,k16 [B*T,64] f16   and  vT [B,64,T] f16 (transposed for PV B-frags).
// One wave = 16 rows x 64 cols, K-loop over C in 32-chunks, 12 WMMAs / iter.
// x is streamed with non-temporal loads (read-once, don't pollute L2).
// ---------------------------------------------------------------------------
__global__ __launch_bounds__(256) void qkv_proj_kernel(
    const float* __restrict__ x, const _Float16* __restrict__ w16,
    _Float16* __restrict__ q16, _Float16* __restrict__ k16, _Float16* __restrict__ vT)
{
  const int lane = threadIdx.x & 31;
  const int wave = threadIdx.x >> 5;
  const int half = lane >> 4;     // 0: lanes 0-15, 1: lanes 16-31
  const int lq   = lane & 15;
  const int rowBase = blockIdx.x * 128 + wave * 16;   // global row in [0, B*T)

  const _Float16* Wq16 = w16;
  const _Float16* Wk16 = w16 + H_ * C_;
  const _Float16* Wv16 = w16 + 2 * H_ * C_;

  const float* xr = x + (size_t)(rowBase + lq) * C_;  // A row = lane%16

  const v8f z = {0.f,0.f,0.f,0.f,0.f,0.f,0.f,0.f};
  v8f accQ[4], accK[4], accV[4];
#pragma unroll
  for (int j = 0; j < 4; ++j) { accQ[j] = z; accK[j] = z; accV[j] = z; }

  for (int kc = 0; kc < C_; kc += 32) {
    // A-frag (x tile 16x32): lane holds K = half*16 + [0..15]
    v8f xlo = __builtin_nontemporal_load((const v8f*)(xr + kc + half * 16));
    v8f xhi = __builtin_nontemporal_load((const v8f*)(xr + kc + half * 16 + 8));
    v16h a = cvt2(xlo, xhi);
#pragma unroll
    for (int j = 0; j < 4; ++j) {
      const int n  = j * 16 + lq;           // output column (head dim)
      const int kb = kc + half * 16;        // B-frag: lane holds K = half*16 + [0..15]
      v16h bq = *(const v16h*)(Wq16 + (size_t)n * C_ + kb);
      v16h bk = *(const v16h*)(Wk16 + (size_t)n * C_ + kb);
      v16h bv = *(const v16h*)(Wv16 + (size_t)n * C_ + kb);
      accQ[j] = __builtin_amdgcn_wmma_f32_16x16x32_f16(false, a, false, bq, (short)0, accQ[j], false, false);
      accK[j] = __builtin_amdgcn_wmma_f32_16x16x32_f16(false, a, false, bk, (short)0, accK[j], false, false);
      accV[j] = __builtin_amdgcn_wmma_f32_16x16x32_f16(false, a, false, bv, (short)0, accV[j], false, false);
    }
  }

  const int b   = rowBase >> 11;      // / T_
  const int tIn = rowBase & (T_ - 1);
#pragma unroll
  for (int j = 0; j < 4; ++j) {
#pragma unroll
    for (int g = 0; g < 8; ++g) {
      const int M = g + half * 8;                       // C/D row for VGPR g
      const int h = j * 16 + lq;                        // C/D col = lane%16
      q16[(size_t)(rowBase + M) * H_ + h] = (_Float16)accQ[j][g];
      k16[(size_t)(rowBase + M) * H_ + h] = (_Float16)accK[j][g];
      vT[((size_t)(b * H_ + h)) * T_ + (tIn + M)] = (_Float16)accV[j][g];
    }
  }
}

// ---------------------------------------------------------------------------
// Phase 2: flash attention.  One wave owns a 16-row q tile; loops s in 32-wide
// chunks: 4 WMMAs for S = q.k^T, online softmax (shfl-xor row reductions in
// the owning 16-lane half), P bounced through LDS (D-layout -> A-layout),
// 4 WMMAs for O += P.V.  4 waves / block; no score matrix ever materialized.
// ---------------------------------------------------------------------------
__global__ __launch_bounds__(128) void attn_kernel(
    const _Float16* __restrict__ q16, const _Float16* __restrict__ k16,
    const _Float16* __restrict__ vT, float* __restrict__ out)
{
  __shared__ __align__(32) _Float16 Pl[4][16 * 32];   // per-wave P bounce buffer

  const int lane = threadIdx.x & 31;
  const int wave = threadIdx.x >> 5;
  const int half = lane >> 4;
  const int lq   = lane & 15;

  const int tile  = blockIdx.x * 4 + wave;   // [0, B*T/16)
  const int b     = tile >> 7;               // / (T_/16)
  const int tBase = (tile & 127) * 16;

  _Float16* Pw = &Pl[wave][0];
  const float scale = 0.03125f;              // C_^-0.5 = 1/32

  // q A-fragments, K = 64 in two 32-chunks (loaded once, reused 64 times)
  v16h qa[2];
#pragma unroll
  for (int hc = 0; hc < 2; ++hc) {
    const _Float16* p = q16 + (size_t)(b * T_ + tBase + lq) * H_ + hc * 32 + half * 16;
    qa[hc] = combine8(*(const v8h*)p, *(const v8h*)(p + 8));
  }

  const v8f z = {0.f,0.f,0.f,0.f,0.f,0.f,0.f,0.f};
  v8f accO[4];
  float mrun[8], lrun[8];
#pragma unroll
  for (int j = 0; j < 4; ++j) accO[j] = z;
#pragma unroll
  for (int g = 0; g < 8; ++g) { mrun[g] = -1.0e30f; lrun[g] = 0.0f; }

  for (int st = 0; st < T_ / 32; ++st) {
    const int sBase = st * 32;

    // ---- scores S = q . k^T for two 16-wide s tiles -----------------------
    v8f accS[2];
#pragma unroll
    for (int nt = 0; nt < 2; ++nt) {
      const int scol = sBase + nt * 16 + lq;            // B-frag col = lane%16
      const _Float16* kp = k16 + (size_t)(b * T_ + scol) * H_ + half * 16;
      v16h kb0 = *(const v16h*)kp;                      // h = 0*32 + half*16 + i
      v16h kb1 = *(const v16h*)(kp + 32);               // h = 1*32 + half*16 + i
      v8f s = __builtin_amdgcn_wmma_f32_16x16x32_f16(false, qa[0], false, kb0, (short)0, z, false, false);
      accS[nt] = __builtin_amdgcn_wmma_f32_16x16x32_f16(false, qa[1], false, kb1, (short)0, s, false, false);
    }

    // ---- online softmax over the 32 new columns ---------------------------
    float p0[8], p1[8], corr[8];
#pragma unroll
    for (int g = 0; g < 8; ++g) {
      float a0 = accS[0][g] * scale;
      float a1 = accS[1][g] * scale;
      float m = fmaxf(a0, a1);
      m = fmaxf(m, __shfl_xor(m, 1, 32));
      m = fmaxf(m, __shfl_xor(m, 2, 32));
      m = fmaxf(m, __shfl_xor(m, 4, 32));
      m = fmaxf(m, __shfl_xor(m, 8, 32));
      const float mn = fmaxf(mrun[g], m);
      corr[g] = __expf(mrun[g] - mn);
      p0[g]   = __expf(a0 - mn);
      p1[g]   = __expf(a1 - mn);
      float rs = p0[g] + p1[g];
      rs += __shfl_xor(rs, 1, 32);
      rs += __shfl_xor(rs, 2, 32);
      rs += __shfl_xor(rs, 4, 32);
      rs += __shfl_xor(rs, 8, 32);
      lrun[g] = lrun[g] * corr[g] + rs;
      mrun[g] = mn;
    }

    // ---- P: D-layout -> LDS -> A-layout; rescale O ------------------------
#pragma unroll
    for (int g = 0; g < 8; ++g) {
      const int M = g + half * 8;
      Pw[M * 32 + lq]      = (_Float16)p0[g];
      Pw[M * 32 + 16 + lq] = (_Float16)p1[g];
    }
#pragma unroll
    for (int j = 0; j < 4; ++j)
#pragma unroll
      for (int g = 0; g < 8; ++g) accO[j][g] *= corr[g];

    asm volatile("s_wait_dscnt 0" ::: "memory");   // intra-wave LDS RAW fence
    v16h pa = combine8(*(const v8h*)&Pw[lq * 32 + half * 16],
                       *(const v8h*)&Pw[lq * 32 + half * 16 + 8]);

    // ---- O += P . V -------------------------------------------------------
#pragma unroll
    for (int j = 0; j < 4; ++j) {
      const _Float16* vp = vT + (size_t)(b * H_ + j * 16 + lq) * T_ + sBase + half * 16;
      v16h vb = *(const v16h*)vp;
      accO[j] = __builtin_amdgcn_wmma_f32_16x16x32_f16(false, pa, false, vb, (short)0, accO[j], false, false);
    }
  }

  // ---- epilogue: out = O / l  (write-once -> non-temporal stores) ---------
#pragma unroll
  for (int j = 0; j < 4; ++j) {
#pragma unroll
    for (int g = 0; g < 8; ++g) {
      const int M = g + half * 8;
      float r = accO[j][g] / lrun[g];
      __builtin_nontemporal_store(
          r, out + (size_t)(b * T_ + tBase + M) * H_ + j * 16 + lq);
    }
  }
}

extern "C" void kernel_launch(void* const* d_in, const int* in_sizes, int n_in,
                              void* d_out, int out_size, void* d_ws, size_t ws_size,
                              hipStream_t stream) {
  (void)in_sizes; (void)n_in; (void)out_size; (void)ws_size;
  const float* x  = (const float*)d_in[0];
  const float* Wk = (const float*)d_in[1];
  const float* Wq = (const float*)d_in[2];
  const float* Wv = (const float*)d_in[3];
  float* out = (float*)d_out;

  _Float16* q16 = (_Float16*)d_ws;                    // [B*T,64] f16  (2 MB)
  _Float16* k16 = q16 + (size_t)B_ * T_ * H_;         // [B*T,64] f16  (2 MB)
  _Float16* vT  = k16 + (size_t)B_ * T_ * H_;         // [B,64,T] f16  (2 MB)
  _Float16* w16 = vT  + (size_t)B_ * T_ * H_;         // 3 x [64,1024] f16 (384 KB)

  cvt_w_kernel<<<(H_ * C_) / 256, 256, 0, stream>>>(Wk, Wq, Wv, w16);
  qkv_proj_kernel<<<(B_ * T_) / 128, 256, 0, stream>>>(x, w16, q16, k16, vT);
  attn_kernel<<<(B_ * T_ / 16) / 4, 128, 0, stream>>>(q16, k16, vT, out);
}
